// GraphTCNAttnQuantileForecaster_22686017258244
// MI455X (gfx1250) — compile-verified, matrix operand order checked by reading, not run on hardware
//
#include <hip/hip_runtime.h>
#include <math.h>

typedef __attribute__((ext_vector_type(16))) _Float16 v16h;
typedef __attribute__((ext_vector_type(8)))  _Float16 v8h;
typedef __attribute__((ext_vector_type(8)))  float    v8f;

#define B_  64
#define T_  96
#define N_  32
#define I_  32
#define S_  16
#define E_  8
#define H_  96
#define BT_  (B_ * T_)        // 6144
#define BTN_ (B_ * T_ * N_)   // 196608

__device__ __forceinline__ float gelu_exact(float x) {
    return 0.5f * x * (1.0f + erff(x * 0.7071067811865475f));
}

// ---------------------------------------------------------------------------
// Generic WMMA GEMM: C[M x 96] = act( A[M x K] @ W[K x 96] + bias + aux )
//   A row-major (lda == K), W row-major (K x 96), K % 32 == 0, M % 16 == 0.
//   auxmode 0: none
//   auxmode 1: + aux[row*96 + col]                          (residual add)
//   auxmode 2: + aux[((row/adiv)*amod + row%amod)*96+col]   (pe broadcast)
//   auxmode 3: + aux[(row/adiv)*96 + col]                   (per-(b,t) bcast)
//   act 1: exact GELU
//
// 8 waves / 256-thread block; wave w owns rows [blk*128 + w*16), all 6 column
// tiles. Weights are pre-swizzled into LDS in B-FRAGMENT order:
//   Wf[((kb*6 + c)*2 + hi)*256 + lane*8 + j]  (halves)
// so each lane reads its 16 B halves as two contiguous 16-byte LDS loads.
// Fragment element e of lane (half = lane>>4, n = lane&15) maps to
//   kloc = (e>>3)*16 + half*8 + (e&7), col = c*16 + n
// which also means each lane's A data is two contiguous 8-float runs.
// ---------------------------------------------------------------------------
__global__ void gemm96_wmma(const float* __restrict__ A,
                            const float* __restrict__ W,
                            const float* __restrict__ bias,
                            const float* __restrict__ aux,
                            float* __restrict__ C,
                            int M, int K,
                            int auxmode, int adiv, int amod, int act)
{
    extern __shared__ _Float16 Wf[];   // (K/32) * 6 * 512 halves
    const int tid = threadIdx.x;
    const int nk  = K >> 5;

    // cooperative fill: swizzle W (K x 96, fp32) -> B-fragment layout (f16)
    for (int i = tid; i < nk * 6 * 512; i += blockDim.x) {
        const int j   = i & 7;
        const int ln  = (i >> 3) & 31;
        const int hi  = (i >> 8) & 1;
        const int c   = (i >> 9) % 6;
        const int kb  = i / (6 * 512);
        const int e     = hi * 8 + j;
        const int half  = ln >> 4;
        const int n     = ln & 15;
        const int kloc  = ((e >> 3) << 4) + (half << 3) + (e & 7);
        Wf[i] = (_Float16)W[(size_t)(kb * 32 + kloc) * 96 + c * 16 + n];
    }
    __syncthreads();

    const int wave = tid >> 5;
    const int lane = tid & 31;
    const int row0 = blockIdx.x * 128 + wave * 16;
    if (row0 >= M) return;              // wave-uniform: EXEC stays all-ones

    const int half = lane >> 4;         // 0 or 1
    const int mr   = lane & 15;

    v8f acc[6];
    #pragma unroll
    for (int c = 0; c < 6; ++c) acc[c] = (v8f){0,0,0,0,0,0,0,0};

    for (int kb = 0; kb < nk; ++kb) {
        // A fragment: two contiguous 8-float runs per lane -> 4x b128 loads
        const float* Arow = A + (size_t)(row0 + mr) * K + kb * 32 + (half << 3);
        const float4 r0 = *(const float4*)(Arow);
        const float4 r1 = *(const float4*)(Arow + 4);
        const float4 r2 = *(const float4*)(Arow + 16);
        const float4 r3 = *(const float4*)(Arow + 20);
        v16h af;
        af[0]  = (_Float16)r0.x;  af[1]  = (_Float16)r0.y;
        af[2]  = (_Float16)r0.z;  af[3]  = (_Float16)r0.w;
        af[4]  = (_Float16)r1.x;  af[5]  = (_Float16)r1.y;
        af[6]  = (_Float16)r1.z;  af[7]  = (_Float16)r1.w;
        af[8]  = (_Float16)r2.x;  af[9]  = (_Float16)r2.y;
        af[10] = (_Float16)r2.z;  af[11] = (_Float16)r2.w;
        af[12] = (_Float16)r3.x;  af[13] = (_Float16)r3.y;
        af[14] = (_Float16)r3.z;  af[15] = (_Float16)r3.w;

        #pragma unroll
        for (int c = 0; c < 6; ++c) {
            const _Float16* bp = Wf + (size_t)((kb * 6 + c) * 2) * 256 + lane * 8;
            const v8h b0 = *(const v8h*)(bp);
            const v8h b1 = *(const v8h*)(bp + 256);
            const v16h bf = __builtin_shufflevector(b0, b1,
                0, 1, 2, 3, 4, 5, 6, 7, 8, 9, 10, 11, 12, 13, 14, 15);
            acc[c] = __builtin_amdgcn_wmma_f32_16x16x32_f16(
                false, af, false, bf, (short)0, acc[c], false, false);
        }
    }

    // Epilogue: C/D layout — VGPR r: M = r + 8*half, N = lane&15
    #pragma unroll
    for (int c = 0; c < 6; ++c) {
        #pragma unroll
        for (int r = 0; r < 8; ++r) {
            const int row = row0 + r + 8 * half;
            const int col = c * 16 + mr;
            float x = acc[c][r];
            if (bias) x += bias[col];
            if (auxmode == 1)      x += aux[(size_t)row * 96 + col];
            else if (auxmode == 2) x += aux[(size_t)((row / adiv) * amod + (row % amod)) * 96 + col];
            else if (auxmode == 3) x += aux[(size_t)(row / adiv) * 96 + col];
            if (act) x = gelu_exact(x);
            C[(size_t)row * 96 + col] = x;
        }
    }
}

// ---------------------------------------------------------------------------
// Weight fusion for step 1: Weff[i,h] = sum_j Wp[i,j]*Wn1[j,h] (i<32),
// cvec[h] = sum_j bp[j]*Wn1[j,h]  (row i==32)
// ---------------------------------------------------------------------------
__global__ void weff_kernel(const float* __restrict__ Wp,
                            const float* __restrict__ bp,
                            const float* __restrict__ Wn1,
                            float* __restrict__ Weff,
                            float* __restrict__ cvec)
{
    int idx = blockIdx.x * blockDim.x + threadIdx.x;
    if (idx >= 33 * 96) return;
    int h = idx % 96, i = idx / 96;
    float acc = 0.f;
    if (i < 32) {
        for (int j = 0; j < 96; ++j) acc += Wp[i * 96 + j] * Wn1[j * 96 + h];
        Weff[i * 96 + h] = acc;
    } else {
        for (int j = 0; j < 96; ++j) acc += bp[j] * Wn1[j * 96 + h];
        cvec[h] = acc;
    }
}

// pe[(b*N+n), h] = bn[h] + cvec[h] + static@Wn_s + edge@Wn_e
__global__ void pe_kernel(const float* __restrict__ nstat,
                          const float* __restrict__ edge,
                          const float* __restrict__ Wn,
                          const float* __restrict__ bn,
                          const float* __restrict__ cvec,
                          float* __restrict__ pe, int s_off, int e_off)
{
    int idx = blockIdx.x * blockDim.x + threadIdx.x;
    if (idx >= B_ * N_ * 96) return;
    int h = idx % 96, bn_i = idx / 96;
    float acc = bn[h] + (cvec ? cvec[h] : 0.f);
    const float* st = nstat + (size_t)bn_i * S_;
    const float* ed = edge  + (size_t)bn_i * E_;
    for (int s = 0; s < S_; ++s) acc += st[s] * Wn[(s_off + s) * 96 + h];
    for (int e = 0; e < E_; ++e) acc += ed[e] * Wn[(e_off + e) * 96 + h];
    pe[idx] = acc;
}

// s[row] = hbuf[row,:] . w2 + b2     (one wave32 per row, 3 elems/lane)
__global__ void score_dot(const float* __restrict__ hbuf,
                          const float* __restrict__ w2,
                          const float* __restrict__ b2,
                          float* __restrict__ s)
{
    int wv   = (blockIdx.x * blockDim.x + threadIdx.x) >> 5;
    int lane = threadIdx.x & 31;
    if (wv >= BTN_) return;
    const float* hr = hbuf + (size_t)wv * 96;
    float acc = hr[lane] * w2[lane] + hr[lane + 32] * w2[lane + 32]
              + hr[lane + 64] * w2[lane + 64];
    for (int off = 16; off; off >>= 1) acc += __shfl_xor(acc, off, 32);
    if (lane == 0) s[wv] = acc + b2[0];
}

// Masked softmax over N=32 neighbors: one wave per (b,t), lane = n.
__global__ void softmax_n(const float* __restrict__ s,
                          const int* __restrict__ mask,
                          float* __restrict__ w)
{
    int wv   = (blockIdx.x * blockDim.x + threadIdx.x) >> 5;
    int lane = threadIdx.x & 31;
    if (wv >= BT_) return;
    int b = wv / T_;
    int m = mask[b * N_ + lane];
    unsigned long long anyb = __ballot(m != 0);
    float x = m ? s[(size_t)wv * N_ + lane] : -__builtin_inff();
    float mx = x;
    for (int off = 16; off; off >>= 1) mx = fmaxf(mx, __shfl_xor(mx, off, 32));
    float e = (m && anyb) ? __expf(x - mx) : 0.f;
    float sum = e;
    for (int off = 16; off; off >>= 1) sum += __shfl_xor(sum, off, 32);
    w[(size_t)wv * N_ + lane] = (anyb != 0ull && sum > 0.f) ? e / sum : 0.f;
}

// agg[bt,h] = sum_n w[bt,n] * ne[bt,n,h]
__global__ void agg_kernel(const float* __restrict__ w,
                           const float* __restrict__ ne,
                           float* __restrict__ agg)
{
    int idx = blockIdx.x * blockDim.x + threadIdx.x;
    if (idx >= BT_ * 96) return;
    int h = idx % 96; size_t bt = idx / 96;
    const float* wr = w  + bt * N_;
    const float* nr = ne + bt * (size_t)N_ * 96 + h;
    float acc = 0.f;
    for (int n = 0; n < N_; ++n) acc += wr[n] * nr[(size_t)n * 96];
    agg[idx] = acc;
}

// cat[r, 0:96] = a[r,:], cat[r, 96:192] = b[r,:]
__global__ void pack2(const float* __restrict__ a, const float* __restrict__ b,
                      float* __restrict__ cat, int rows)
{
    int idx = blockIdx.x * blockDim.x + threadIdx.x;
    if (idx >= rows * 192) return;
    int c = idx % 192, r = idx / 192;
    cat[idx] = (c < 96) ? a[(size_t)r * 96 + c] : b[(size_t)r * 96 + (c - 96)];
}

// y = x + gelu(causal dilated conv(x));  x,y (B,T,H); w (O,I,3)
__global__ void tcn_kernel(const float* __restrict__ x,
                           const float* __restrict__ w,
                           const float* __restrict__ bias,
                           float* __restrict__ y, int d)
{
    int idx = blockIdx.x * blockDim.x + threadIdx.x;
    if (idx >= BT_ * 96) return;
    int o = idx % 96, bt = idx / 96, t = bt % T_, b = bt / T_;
    float acc = bias[o];
    for (int k = 0; k < 3; ++k) {
        int tt = t - (2 - k) * d;
        if (tt < 0) continue;
        const float* xr = x + ((size_t)b * T_ + tt) * 96;
        const float* wr = w + (size_t)o * 96 * 3 + k;
        for (int i = 0; i < 96; ++i) acc += wr[i * 3] * xr[i];
    }
    y[idx] = x[idx] + gelu_exact(acc);
}

// Temporal attention pool: one block per b.
__global__ void pool_kernel(const float* __restrict__ enc,
                            const float* __restrict__ wp,
                            const float* __restrict__ bp,
                            float* __restrict__ tfeat)
{
    __shared__ float sc[T_];
    __shared__ float mx, sm;
    int b = blockIdx.x, tid = threadIdx.x;
    if (tid < T_) {
        const float* er = enc + ((size_t)b * T_ + tid) * 96;
        float a = 0.f;
        for (int h = 0; h < 96; ++h) a += er[h] * wp[h];
        sc[tid] = a + bp[0];
    }
    __syncthreads();
    if (tid == 0) { float m = -__builtin_inff(); for (int t = 0; t < T_; ++t) m = fmaxf(m, sc[t]); mx = m; }
    __syncthreads();
    if (tid < T_) sc[tid] = __expf(sc[tid] - mx);
    __syncthreads();
    if (tid == 0) { float s2 = 0.f; for (int t = 0; t < T_; ++t) s2 += sc[t]; sm = s2; }
    __syncthreads();
    if (tid < 96) {
        float a = 0.f;
        for (int t = 0; t < T_; ++t) a += sc[t] * enc[((size_t)b * T_ + t) * 96 + tid];
        tfeat[(size_t)b * 96 + tid] = a / sm;
    }
}

// static MLP: gelu(gelu(s@W0+b0)@W1+b1); one block per b.
__global__ void static_mlp_kernel(const float* __restrict__ st,
                                  const float* __restrict__ W0, const float* __restrict__ b0,
                                  const float* __restrict__ W1, const float* __restrict__ b1,
                                  float* __restrict__ sf)
{
    __shared__ float h1[96];
    int b = blockIdx.x, tid = threadIdx.x;
    if (tid < 96) {
        float a = b0[tid];
        for (int s = 0; s < S_; ++s) a += st[b * S_ + s] * W0[s * 96 + tid];
        h1[tid] = gelu_exact(a);
    }
    __syncthreads();
    if (tid < 96) {
        float a = b1[tid];
        for (int j = 0; j < 96; ++j) a += h1[j] * W1[j * 96 + tid];
        sf[(size_t)b * 96 + tid] = gelu_exact(a);
    }
}

// head1 (96 -> 9) + per-quantile 3-element sort; one thread per b.
__global__ void head_final(const float* __restrict__ hh,
                           const float* __restrict__ W,
                           const float* __restrict__ bias,
                           float* __restrict__ out)
{
    int b = blockIdx.x * blockDim.x + threadIdx.x;
    if (b >= B_) return;
    float x[9];
    for (int j = 0; j < 9; ++j) {
        float a = bias[j];
        for (int h = 0; h < 96; ++h) a += hh[(size_t)b * 96 + h] * W[h * 9 + j];
        x[j] = a;
    }
    for (int q = 0; q < 3; ++q) {
        float a = x[3 * q], c = x[3 * q + 1], d = x[3 * q + 2];
        float lo = fminf(a, fminf(c, d));
        float hi = fmaxf(a, fmaxf(c, d));
        out[(size_t)b * 9 + 3 * q]     = lo;
        out[(size_t)b * 9 + 3 * q + 1] = a + c + d - lo - hi;
        out[(size_t)b * 9 + 3 * q + 2] = hi;
    }
}

// ---------------------------------------------------------------------------
extern "C" void kernel_launch(void* const* d_in, const int* in_sizes, int n_in,
                              void* d_out, int out_size, void* d_ws, size_t ws_size,
                              hipStream_t stream)
{
    (void)in_sizes; (void)n_in; (void)out_size; (void)ws_size;
    const float* seq   = (const float*)d_in[0];
    const float* stat  = (const float*)d_in[1];
    const float* nseq  = (const float*)d_in[2];
    const float* nstat = (const float*)d_in[3];
    const float* edge  = (const float*)d_in[4];
    const int*   mask  = (const int*)d_in[5];
    // JAX pytree flatten order (dict keys sorted, 'b' < 'w'):
    // aggregators[0..1]{center, neighbor, out1, out2, score1, score2}{b,w},
    // head[0..1]{b,w}, neighbor_seq_proj{b,w}, pool{b,w},
    // static_mlp[0..1]{b,w}, tcn[0..1]{b,w}
    const float* const* P = (const float* const*)(d_in + 6);
    const float* head0_b = P[24]; const float* head0_w = P[25];
    const float* head1_b = P[26]; const float* head1_w = P[27];
    const float* proj_b  = P[28]; const float* proj_w  = P[29];
    const float* pool_b  = P[30]; const float* pool_w  = P[31];
    const float* sm0_b   = P[32]; const float* sm0_w   = P[33];
    const float* sm1_b   = P[34]; const float* sm1_w   = P[35];
    const float* tcn0_b  = P[36]; const float* tcn0_w  = P[37];
    const float* tcn1_b  = P[38]; const float* tcn1_w  = P[39];

    // workspace layout (floats)
    float* wsf = (float*)d_ws;
    size_t off = 0;
    auto alloc = [&](size_t n) { float* p = wsf + off; off += n; return p; };
    float* ne      = alloc((size_t)BTN_ * 96);
    float* hbuf    = alloc((size_t)BTN_ * 96);
    float* ce      = alloc((size_t)BT_ * 96);
    float* s1ce    = alloc((size_t)BT_ * 96);
    float* center  = alloc((size_t)BT_ * 96);
    float* center2 = alloc((size_t)BT_ * 96);
    float* aggb    = alloc((size_t)BT_ * 96);
    float* catb    = alloc((size_t)BT_ * 192);
    float* g1      = alloc((size_t)BT_ * 96);
    float* sbuf    = alloc((size_t)BTN_);
    float* wbuf    = alloc((size_t)BTN_);
    float* pe      = alloc((size_t)B_ * N_ * 96);
    float* weff    = alloc(32 * 96);
    float* cvec    = alloc(96);
    float* enc1    = alloc((size_t)BT_ * 96);
    float* enc2    = alloc((size_t)BT_ * 96);
    float* tfeat   = alloc((size_t)B_ * 96);
    float* sf      = alloc((size_t)B_ * 96);
    float* cat64   = alloc((size_t)B_ * 192);
    float* hh      = alloc((size_t)B_ * 96);

    auto cdiv = [](int a, int b) { return (a + b - 1) / b; };
    auto gemm = [&](const float* A, const float* W, const float* bias,
                    const float* aux, float* C, int M, int K,
                    int auxmode, int adiv, int amod, int act) {
        const size_t shmem = (size_t)(K / 32) * 6 * 512 * sizeof(_Float16);
        gemm96_wmma<<<cdiv(M, 128), 256, shmem, stream>>>(
            A, W, bias, aux, C, M, K, auxmode, adiv, amod, act);
    };

    // fold neighbor_seq_proj into step-1 neighbor weights
    weff_kernel<<<cdiv(33 * 96, 256), 256, 0, stream>>>(proj_w, proj_b, P[15], weff, cvec);

    for (int k = 0; k < 2; ++k) {
        const float* bc  = P[k * 12 + 0]; const float* Wc  = P[k * 12 + 1];
        const float* bn  = P[k * 12 + 2]; const float* Wn  = P[k * 12 + 3];
        const float* bo1 = P[k * 12 + 4]; const float* Wo1 = P[k * 12 + 5];
        const float* bo2 = P[k * 12 + 6]; const float* Wo2 = P[k * 12 + 7];
        const float* bs1 = P[k * 12 + 8]; const float* Ws1 = P[k * 12 + 9];
        const float* bs2 = P[k * 12 + 10]; const float* Ws2 = P[k * 12 + 11];

        // pe: broadcast (static, edge) parts of the neighbor linear (+cvec for k=1)
        pe_kernel<<<cdiv(B_ * N_ * 96, 256), 256, 0, stream>>>(
            nstat, edge, Wn, bn, (k ? cvec : nullptr), pe,
            (k ? 96 : 32), (k ? 112 : 48));
        // ne = nseq @ Wd(32x96) + pe     (pe broadcast over t)
        gemm(nseq, (k ? weff : Wn), nullptr, pe, ne, BTN_, 32, 2, T_ * N_, N_, 0);
        // ce = center @ Wc + bc
        gemm((k ? center : seq), Wc, bc, nullptr, ce, BT_, (k ? 96 : 32), 0, 1, 1, 0);
        // s1ce = ce @ Ws1[0:96] + bs1
        gemm(ce, Ws1, bs1, nullptr, s1ce, BT_, 96, 0, 1, 1, 0);
        // hbuf = gelu(ne @ Ws1[96:192] + s1ce)   (s1ce broadcast over n)
        gemm(ne, Ws1 + 96 * 96, nullptr, s1ce, hbuf, BTN_, 96, 3, N_, 1, 1);
        // s = hbuf . w2 + b2 ; softmax over n ; agg
        score_dot<<<cdiv(BTN_ * 32, 256), 256, 0, stream>>>(hbuf, Ws2, bs2, sbuf);
        softmax_n<<<cdiv(BT_ * 32, 256), 256, 0, stream>>>(sbuf, mask, wbuf);
        agg_kernel<<<cdiv(BT_ * 96, 256), 256, 0, stream>>>(wbuf, ne, aggb);
        // out path: cat(ce, agg) -> gelu(out1) -> out2 + ce (residual)
        pack2<<<cdiv(BT_ * 192, 256), 256, 0, stream>>>(ce, aggb, catb, BT_);
        gemm(catb, Wo1, bo1, nullptr, g1, BT_, 192, 0, 1, 1, 1);
        gemm(g1, Wo2, bo2, ce, (k ? center2 : center), BT_, 96, 1, 1, 1, 0);
    }

    // TCN encoder: two dilated causal convs with GELU + residual
    tcn_kernel<<<cdiv(BT_ * 96, 256), 256, 0, stream>>>(center2, tcn0_w, tcn0_b, enc1, 1);
    tcn_kernel<<<cdiv(BT_ * 96, 256), 256, 0, stream>>>(enc1, tcn1_w, tcn1_b, enc2, 2);
    // temporal attention pool, static MLP, head
    pool_kernel<<<B_, 128, 0, stream>>>(enc2, pool_w, pool_b, tfeat);
    static_mlp_kernel<<<B_, 128, 0, stream>>>(stat, sm0_w, sm0_b, sm1_w, sm1_b, sf);
    pack2<<<cdiv(B_ * 192, 256), 256, 0, stream>>>(tfeat, sf, cat64, B_);
    gemm(cat64, head0_w, head0_b, nullptr, hh, B_, 192, 0, 1, 1, 1);
    head_final<<<1, 64, 0, stream>>>(hh, head1_w, head1_b, (float*)d_out);
}
